// Bottleneck_75325136437765
// MI455X (gfx1250) — matrix operand order repaired and checked by reading, not compiled, not run on hardware
//
#include <hip/hip_runtime.h>
#include <hip/hip_bf16.h>

typedef __attribute__((ext_vector_type(16))) __bf16 v16bf;
typedef __attribute__((ext_vector_type(8)))  __bf16 v8bf;
typedef __attribute__((ext_vector_type(4)))  __bf16 v4bf;
typedef __attribute__((ext_vector_type(8)))  float  v8f;

#define HH 32
#define WW 32
#define NPIX (HH * WW)          // 1024 pixels per image
#define NB 16
#define MTOT (NB * NPIX)        // 16384 global pixel rows
#define CMID 256
#define COUT 1024
#define CIN  1024
#define NHEADS 32
#define DHEAD 8

// ---------------------------------------------------------------------------
// Flat fp32 -> bf16 convert (weights). n4 = n/4.
// ---------------------------------------------------------------------------
__global__ __launch_bounds__(256)
void cvt_f32_bf16(const float* __restrict__ src, __bf16* __restrict__ dst, int n4)
{
    const int i = blockIdx.x * 256 + threadIdx.x;
    if (i < n4) {
        const float4 f = ((const float4*)src)[i];
        v4bf o;
        o[0] = (__bf16)f.x; o[1] = (__bf16)f.y;
        o[2] = (__bf16)f.z; o[3] = (__bf16)f.w;
        ((v4bf*)dst)[i] = o;
    }
}

// ---------------------------------------------------------------------------
// Tiled transpose + fp32->bf16: src (R, C) fp32 -> dst (C, R) bf16.
// Grid (R/32, C/32, batch), block (32, 8). LDS 32x33 avoids bank conflicts.
// Used only for x: per-image (CIN, NPIX) -> (NPIX, CIN).
// ---------------------------------------------------------------------------
__global__ __launch_bounds__(256)
void transpose_cvt(const float* __restrict__ src, __bf16* __restrict__ dst,
                   int R, int C, long src_bstride, long dst_bstride)
{
    __shared__ __bf16 tile[32][33];
    const float* __restrict__ s = src + (size_t)blockIdx.z * src_bstride;
    __bf16* __restrict__ d      = dst + (size_t)blockIdx.z * dst_bstride;
    const int r0 = blockIdx.x << 5;
    const int c0 = blockIdx.y << 5;
    const int tx = threadIdx.x;
    const int ty = threadIdx.y;
#pragma unroll
    for (int i = 0; i < 4; ++i) {
        const int r = (ty << 2) + i;
        tile[r][tx] = (__bf16)s[(size_t)(r0 + r) * C + c0 + tx];
    }
    __syncthreads();
#pragma unroll
    for (int i = 0; i < 4; ++i) {
        const int c = (ty << 2) + i;
        d[(size_t)(c0 + c) * R + r0 + tx] = tile[tx][c];
    }
}

// ---------------------------------------------------------------------------
// NHWC GEMM: Y[pix, n] = act( scale[n] * (sum_k XT[pix,k]*Wb[n,k]) + bias[n] )
// XT bf16 (MTOT, K) pixel-major; Wb bf16 (N, K) ROW-MAJOR (original layout!):
// the WMMA B fragment for lane (hi,lo) is Wb[nb+lo][kk+16hi .. +15] -- 16
// contiguous bf16 = one aligned 32B vector load. A fragment = 2x b128.
// Wave tile = 128(M) x 16(N); phase-split loads -> WMMAs for latency overlap.
// Grid (MTOT/128, N/128), block 256 (8 waves split N).
// out_f32=0: Y bf16 (MTOT, N).  out_f32=1: Y fp32 NCHW (NB, N, NPIX) + resid.
// ---------------------------------------------------------------------------
__global__ __launch_bounds__(256)
void gemm_nhwc_wmma(const __bf16* __restrict__ XT, const __bf16* __restrict__ Wb,
                    void* __restrict__ Yv,
                    const float* __restrict__ scale, const float* __restrict__ bias,
                    const float* __restrict__ resid,
                    int N, int K, int do_relu, int out_f32)
{
    const int wave  = threadIdx.x >> 5;
    const int lane  = threadIdx.x & 31;
    const int hi    = lane >> 4;
    const int lo    = lane & 15;
    const int mbase = blockIdx.x << 7;                     // 128 pixel rows
    const int nb    = (blockIdx.y << 7) + (wave << 4);     // 16 channels/wave
    const int ch    = nb + lo;

    v8f acc[8];
#pragma unroll
    for (int s = 0; s < 8; ++s) acc[s] = v8f{};

    // A: lane = pixel row, halfs j -> K = kk+16hi+j (16 contiguous)
    const __bf16* __restrict__ Xrow0 = XT + (size_t)(mbase + lo) * K + 16 * hi;
    // B: halfs j -> K = kk+16hi+j, i.e. 16 contiguous elements of W row `ch`
    const __bf16* __restrict__ Wrow  = Wb + (size_t)ch * K + 16 * hi;

    for (int kk = 0; kk < K; kk += 32) {
        const v16bf bf = *(const v16bf*)(Wrow + kk);
        // phase 1: issue all A loads into distinct registers
        v16bf a[8];
#pragma unroll
        for (int s = 0; s < 8; ++s) {
            const __bf16* __restrict__ Ar = Xrow0 + (size_t)(s << 4) * K + kk;
            const v8bf a0 = *(const v8bf*)(Ar);
            const v8bf a1 = *(const v8bf*)(Ar + 8);
            a[s] = __builtin_shufflevector(a0, a1,
                0, 1, 2, 3, 4, 5, 6, 7, 8, 9, 10, 11, 12, 13, 14, 15);
        }
        // phase 2: 8 independent WMMAs
#pragma unroll
        for (int s = 0; s < 8; ++s)
            acc[s] = __builtin_amdgcn_wmma_f32_16x16x32_bf16(
                         false, a[s], false, bf, (short)0, acc[s], false, false);
    }

    // epilogue: channel is wave-invariant per lane -> one scale/bias load
    const float sm = scale[ch];
    const float bm = bias[ch];
#pragma unroll
    for (int s = 0; s < 8; ++s) {
#pragma unroll
        for (int r = 0; r < 8; ++r) {
            const int pglob = mbase + (s << 4) + r + 8 * hi;  // C: VGPR r -> M
            float vv = acc[s][r] * sm + bm;
            if (out_f32) {
                const int b = pglob >> 10, p = pglob & (NPIX - 1);
                const size_t idx = ((size_t)b * N + ch) * NPIX + p;
                if (resid) vv += resid[idx];
                if (do_relu) vv = fmaxf(vv, 0.f);
                ((float*)Yv)[idx] = vv;
            } else {
                if (do_relu) vv = fmaxf(vv, 0.f);
                ((__bf16*)Yv)[(size_t)pglob * N + ch] = (__bf16)vv;
            }
        }
    }
}

// ---------------------------------------------------------------------------
// Local 3x3 attention, NHWC bf16: q/k/v/out (MTOT, CMID).
// Thread = (pixel, head); one wave = 32 heads of one pixel (uniform padding
// predicate). All q/k/v taps are 16B vector loads. Grid (MTOT/8), block 256.
// ---------------------------------------------------------------------------
__global__ __launch_bounds__(256)
void attn_kernel(const __bf16* __restrict__ q, const __bf16* __restrict__ k,
                 const __bf16* __restrict__ v,
                 const float* __restrict__ pos_h, const float* __restrict__ pos_w,
                 const float* __restrict__ bnatt_s, const float* __restrict__ bnatt_b,
                 __bf16* __restrict__ out)
{
    const int g     = threadIdx.x & 31;
    const int pglob = (blockIdx.x << 3) + (threadIdx.x >> 5);
    const int b = pglob >> 10, p = pglob & (NPIX - 1);
    const int h = p >> 5, w = p & 31;
    const int c0 = g * DHEAD;

    float qv[DHEAD];
    {
        const v8bf qr = *(const v8bf*)(q + (size_t)pglob * CMID + c0);
#pragma unroll
        for (int e = 0; e < DHEAD; ++e) qv[e] = (float)qr[e];
    }

    float logit[9];
#pragma unroll
    for (int t = 0; t < 9; ++t) {
        const int hn = h + t / 3 - 1, wn = w + t % 3 - 1;
        const bool inb = (hn >= 0) & (hn < HH) & (wn >= 0) & (wn < WW);
        float acc = 0.f;
        if (inb) {
            const int pn = (b << 10) + hn * WW + wn;
            const v8bf kr = *(const v8bf*)(k + (size_t)pn * CMID + c0);
#pragma unroll
            for (int e = 0; e < DHEAD; ++e) {
                const float pe = pos_h[(c0 + e) * 3 + t / 3] + pos_w[(c0 + e) * 3 + t % 3];
                acc += qv[e] * ((float)kr[e] + pe);
            }
        } else {
            // unfold pads k with 0, but pos is still added
#pragma unroll
            for (int e = 0; e < DHEAD; ++e) {
                const float pe = pos_h[(c0 + e) * 3 + t / 3] + pos_w[(c0 + e) * 3 + t % 3];
                acc += qv[e] * pe;
            }
        }
        logit[t] = acc;
    }

    float mx = logit[0];
#pragma unroll
    for (int t = 1; t < 9; ++t) mx = fmaxf(mx, logit[t]);
    float ssum = 0.f;
#pragma unroll
    for (int t = 0; t < 9; ++t) { logit[t] = __expf(logit[t] - mx); ssum += logit[t]; }
    const float inv = 1.f / ssum;

    float o[DHEAD];
#pragma unroll
    for (int e = 0; e < DHEAD; ++e) o[e] = 0.f;
#pragma unroll
    for (int t = 0; t < 9; ++t) {
        const int hn = h + t / 3 - 1, wn = w + t % 3 - 1;
        const bool inb = (hn >= 0) & (hn < HH) & (wn >= 0) & (wn < WW);
        if (inb) {
            const int pn = (b << 10) + hn * WW + wn;
            const v8bf vr = *(const v8bf*)(v + (size_t)pn * CMID + c0);
#pragma unroll
            for (int e = 0; e < DHEAD; ++e) o[e] += (float)vr[e] * logit[t];
        }
    }
    v8bf ob;
#pragma unroll
    for (int e = 0; e < DHEAD; ++e) {
        const float val = o[e] * inv * bnatt_s[c0 + e] + bnatt_b[c0 + e];
        ob[e] = (__bf16)fmaxf(val, 0.f);
    }
    *(v8bf*)(out + (size_t)pglob * CMID + c0) = ob;
}

// ---------------------------------------------------------------------------
extern "C" void kernel_launch(void* const* d_in, const int* in_sizes, int n_in,
                              void* d_out, int out_size, void* d_ws, size_t ws_size,
                              hipStream_t stream)
{
    const float* x       = (const float*)d_in[0];   // (16,1024,32,32)
    const float* w_conv1 = (const float*)d_in[1];   // (256,1024)  = (N,K)
    const float* bn1_s   = (const float*)d_in[2];
    const float* bn1_b   = (const float*)d_in[3];
    const float* wq      = (const float*)d_in[4];   // (256,256)
    const float* bnq_s   = (const float*)d_in[5];
    const float* bnq_b   = (const float*)d_in[6];
    const float* wk      = (const float*)d_in[7];
    const float* bnk_s   = (const float*)d_in[8];
    const float* bnk_b   = (const float*)d_in[9];
    const float* wv      = (const float*)d_in[10];
    const float* bnv_s   = (const float*)d_in[11];
    const float* bnv_b   = (const float*)d_in[12];
    const float* pos_h   = (const float*)d_in[13];  // (256,3,1)
    const float* pos_w   = (const float*)d_in[14];  // (256,1,3)
    const float* bnatt_s = (const float*)d_in[15];
    const float* bnatt_b = (const float*)d_in[16];
    const float* w_conv3 = (const float*)d_in[17];  // (1024,256) = (N,K)
    const float* bn3_s   = (const float*)d_in[18];
    const float* bn3_b   = (const float*)d_in[19];
    float* out = (float*)d_out;                     // (16,1024,32,32) fp32

    // ---- workspace layout (all bf16, 32B-aligned segments) ---------------
    __bf16* xT  = (__bf16*)d_ws;                    // (MTOT, CIN) pixel-major
    __bf16* w1b = xT  + (size_t)MTOT * CIN;         // (256,1024) row-major
    __bf16* wqb = w1b + (size_t)CMID * CIN;         // (256,256)
    __bf16* wkb = wqb + (size_t)CMID * CMID;
    __bf16* wvb = wkb + (size_t)CMID * CMID;
    __bf16* w3b = wvb + (size_t)CMID * CMID;        // (1024,256)
    __bf16* h_buf = w3b + (size_t)COUT * CMID;      // (MTOT, CMID)
    __bf16* q_buf = h_buf + (size_t)MTOT * CMID;
    __bf16* k_buf = q_buf + (size_t)MTOT * CMID;
    __bf16* v_buf = k_buf + (size_t)MTOT * CMID;
    __bf16* att   = h_buf;                          // h dead after q/k/v

    const dim3 blk(256);
    // ---- one-time weight converts (keep original row-major (N,K) layout) -
    auto cvt = [&](const float* s, __bf16* d, size_t n) {
        const int n4 = (int)(n / 4);
        cvt_f32_bf16<<<dim3((n4 + 255) / 256), blk, 0, stream>>>(s, d, n4);
    };
    cvt(w_conv1, w1b, (size_t)CMID * CIN);
    cvt(wq,      wqb, (size_t)CMID * CMID);
    cvt(wk,      wkb, (size_t)CMID * CMID);
    cvt(wv,      wvb, (size_t)CMID * CMID);
    cvt(w_conv3, w3b, (size_t)COUT * CMID);
    // ---- x: per-image (CIN, NPIX) -> (NPIX, CIN), batched over z ---------
    transpose_cvt<<<dim3(CIN / 32, NPIX / 32, NB), dim3(32, 8), 0, stream>>>(
        x, xT, CIN, NPIX, (long)CIN * NPIX, (long)NPIX * CIN);

    // ---- pipeline --------------------------------------------------------
    // conv1: N=256, K=1024
    gemm_nhwc_wmma<<<dim3(MTOT / 128, CMID / 128), blk, 0, stream>>>(
        xT, w1b, h_buf, bn1_s, bn1_b, nullptr, CMID, CIN, 1, 0);
    // q / k / v: N=256, K=256
    gemm_nhwc_wmma<<<dim3(MTOT / 128, CMID / 128), blk, 0, stream>>>(
        h_buf, wqb, q_buf, bnq_s, bnq_b, nullptr, CMID, CMID, 1, 0);
    gemm_nhwc_wmma<<<dim3(MTOT / 128, CMID / 128), blk, 0, stream>>>(
        h_buf, wkb, k_buf, bnk_s, bnk_b, nullptr, CMID, CMID, 1, 0);
    gemm_nhwc_wmma<<<dim3(MTOT / 128, CMID / 128), blk, 0, stream>>>(
        h_buf, wvb, v_buf, bnv_s, bnv_b, nullptr, CMID, CMID, 0, 0);
    // attention -> att (bf16 NHWC, reuses h slot)
    attn_kernel<<<dim3(MTOT / 8), blk, 0, stream>>>(
        q_buf, k_buf, v_buf, pos_h, pos_w, bnatt_s, bnatt_b, att);
    // conv3 + residual + relu: N=1024, K=256, fp32 NCHW out
    gemm_nhwc_wmma<<<dim3(MTOT / 128, COUT / 128), blk, 0, stream>>>(
        att, w3b, out, bn3_s, bn3_b, x, COUT, CMID, 1, 1);
}
